// GGANLSTMModelWithHourlyHeads_31164282699793
// MI455X (gfx1250) — matrix operand order repaired and checked by reading, not compiled, not run on hardware
//
#include <hip/hip_runtime.h>
#include <hip/hip_bf16.h>
#include <math.h>

// ---------------------------------------------------------------------------
// GGAN-LSTM forward for MI455X (gfx1250), wave32 + WMMA bf16.
//
// Roofline: dense GEMM work is ~240 GFLOP, edge gather/scatter moves ~25 GB
// (~1.1 ms at 23.3 TB/s HBM, mostly L2-resident: per-timestep intermediates
// ~165 MB < 192 MB L2). GEMMs therefore run as bf16 x bf16 -> f32 WMMA
// (v_wmma_f32_16x16x32_bf16) to halve operand traffic with fp32 accumulation.
// Each wave computes a 16x64 C tile (4 WMMA accumulators share one A fragment).
// All N dims are padded to multiples of 64 so the inner loop has NO divergent
// guards: WMMA must issue with EXEC all-1s (ISA 7.12), and uniform control
// flow lets the compiler software-pipeline loads against v_wmma.
// ---------------------------------------------------------------------------

typedef unsigned short ushort_t;
typedef __attribute__((ext_vector_type(16))) __bf16 v16bf;
typedef __attribute__((ext_vector_type(8)))  float  v8f;

static constexpr int NN   = 30000;
static constexpr int NE   = 960000;
static constexpr int NG   = 64;
static constexpr int TT   = 12;
static constexpr int NRELC= 5;

#define DEVFN static __device__ __forceinline__

DEVFN ushort_t f2bf(float f){
  union { float f; unsigned u; } v; v.f = f;
  unsigned u = v.u;
  unsigned r = (u + 0x7FFFu + ((u >> 16) & 1u)) >> 16;   // round-to-nearest-even
  return (ushort_t)r;
}
DEVFN float sigm(float x){ return 1.0f / (1.0f + __expf(-x)); }

DEVFN v16bf load_frag16(const ushort_t* p0, const ushort_t* p1){
  union { v16bf v; uint4 q[2]; } u;
  u.q[0] = *reinterpret_cast<const uint4*>(p0);
  u.q[1] = *reinterpret_cast<const uint4*>(p1);
  return u.v;
}

// ---------------------------------------------------------------------------
// WMMA GEMM: C[M x N](f32) = A[M x K](bf16,row-major) * Bt[N x K]^T (+bias)
//   One wave per 16x64 C tile: 4 accumulators reuse a single A fragment.
//   Requires N % 64 == 0 and K % 32 == 0 (host pads). No divergent control
//   flow in the K loop (WMMA requires EXEC all-1s).
//   Fragment layouts (A/B k order is this code's contiguous choice; C/D per ISA 7.12.2):
//     A 16x32: lanes 0-15 -> M rows, K 0..15; lanes 16-31 K 16..31
//     B 32x16: lanes 0-15 -> col, K 0..15; lanes 16-31 -> col, K 16..31
//     C/D:     lane = n + 16*(m/8), vgpr r = m%8
//   flags bit0: accumulate into existing C.
// ---------------------------------------------------------------------------
__global__ void __launch_bounds__(128)
k_gemm(const ushort_t* __restrict__ A, const ushort_t* __restrict__ Bt,
       const float* __restrict__ bias, float* __restrict__ C,
       int M, int N, int K, int flags){
  int wave    = threadIdx.x >> 5;
  int lane    = threadIdx.x & 31;
  int tilesN64= N >> 6;
  int tileId  = blockIdx.x * 4 + wave;
  if (tileId >= (M >> 4) * tilesN64) return;
  int tm  = (tileId / tilesN64) << 4;
  int tn0 = (tileId % tilesN64) << 6;
  int mr  = lane & 15;
  int kh  = lane >> 4;

  const ushort_t* ap = A + (size_t)(tm + mr) * K;

  v8f acc[4] = {{}, {}, {}, {}};
  for (int k0 = 0; k0 < K; k0 += 32){
    __builtin_prefetch(ap + k0 + 256, 0, 3);
    v16bf a = load_frag16(ap + k0 + kh * 16, ap + k0 + kh * 16 + 8);
    #pragma unroll
    for (int j = 0; j < 4; ++j){
      const ushort_t* bp = Bt + (size_t)(tn0 + j * 16 + mr) * K + k0 + kh * 16;
      v16bf b = load_frag16(bp, bp + 8);
      acc[j] = __builtin_amdgcn_wmma_f32_16x16x32_bf16(false, a, false, b,
                                                       (short)0, acc[j], false, false);
    }
  }
  int mb = tm + kh * 8;
  #pragma unroll
  for (int j = 0; j < 4; ++j){
    int n = tn0 + j * 16 + mr;
    float bv = bias ? bias[n] : 0.0f;
    #pragma unroll
    for (int r = 0; r < 8; ++r){
      size_t o = (size_t)(mb + r) * N + n;
      float v = acc[j][r] + bv;
      if (flags & 1) v += C[o];
      C[o] = v;
    }
  }
}

// ------------------------- packing / conversion ----------------------------

// normalize x_seq[t] and convert to bf16 (NN x 32)
__global__ void k_pack_x(const float* __restrict__ x, const float* __restrict__ mean,
                         const float* __restrict__ stdv, ushort_t* __restrict__ out){
  int idx = blockIdx.x * 256 + threadIdx.x;
  if (idx >= NN * 32) return;
  int f = idx & 31;
  out[idx] = f2bf((x[idx] - mean[f]) / (stdv[f] + 1e-8f));
}

// W (K x N) f32 -> Bt (Np x Kp) bf16, zero padded
__global__ void k_pack_wT(const float* __restrict__ W, ushort_t* __restrict__ out,
                          int K, int N, int Kp, int Np){
  int idx = blockIdx.x * 256 + threadIdx.x;
  if (idx >= Np * Kp) return;
  int n = idx / Kp, k = idx % Kp;
  out[idx] = (n < N && k < K) ? f2bf(W[(size_t)k * N + n]) : (ushort_t)0;
}

// root(K x 128) + w_rel(5 x K x 128) -> Bt (768 x K) bf16
__global__ void k_pack_rgcnT(const float* __restrict__ wroot, const float* __restrict__ wrel,
                             ushort_t* __restrict__ out, int K){
  int idx = blockIdx.x * 256 + threadIdx.x;
  if (idx >= 768 * K) return;
  int n = idx / K, k = idx % K;
  float v;
  if (n < 128) v = wroot[(size_t)k * 128 + n];
  else {
    int r = (n - 128) >> 7, c = (n - 128) & 127;
    v = wrel[((size_t)r * K + k) * 128 + c];
  }
  out[idx] = f2bf(v);
}

// heads w1 (12 x 128 x 64) -> 12 x Bt(64 x 128)
__global__ void k_pack_headT(const float* __restrict__ w1, ushort_t* __restrict__ out){
  int idx = blockIdx.x * 256 + threadIdx.x;
  if (idx >= TT * 64 * 128) return;
  int t = idx / 8192, r = idx % 8192;
  int n = r / 128, k = r % 128;
  out[idx] = f2bf(w1[((size_t)t * 128 + k) * 64 + n]);
}

__global__ void k_pad_bias(const float* __restrict__ b, float* __restrict__ out, int N, int Np){
  int idx = blockIdx.x * 256 + threadIdx.x;
  if (idx >= Np) return;
  out[idx] = (idx < N) ? b[idx] : 0.0f;
}

__global__ void k_pack_bf(const float* __restrict__ in, ushort_t* __restrict__ out, int n){
  int idx = blockIdx.x * 256 + threadIdx.x;
  if (idx < n) out[idx] = f2bf(in[idx]);
}

// ------------------------- RGCN message passing ----------------------------

// one wave per edge; gathers 128 f32 (4/lane, float4) from HC (NN x 768 with
// relation slabs at col 128 + r*128) and atomically accumulates per-relation.
__global__ void __launch_bounds__(256)
k_edge_scatter(const float* __restrict__ HC, const int* __restrict__ ei,
               const int* __restrict__ et, float* __restrict__ acc,
               float* __restrict__ cnt){
  int gid  = blockIdx.x * 256 + threadIdx.x;
  int e    = gid >> 5;
  int lane = threadIdx.x & 31;
  if (e >= NE) return;
  int src = ei[e], dst = ei[NE + e], r = et[e];
  const float4 m = *reinterpret_cast<const float4*>(
      HC + (size_t)src * 768 + 128 + (size_t)r * 128 + lane * 4);
  float* a = acc + ((size_t)r * NN + dst) * 128 + lane * 4;
  atomicAdd(a + 0, m.x); atomicAdd(a + 1, m.y);
  atomicAdd(a + 2, m.z); atomicAdd(a + 3, m.w);
  if (lane == 0) atomicAdd(cnt + (size_t)r * NN + dst, 1.0f);
}

// out = PReLU(BN(root + sum_r acc_r/max(cnt_r,1))) (+ residual); write f32+bf16
__global__ void __launch_bounds__(128)
k_rgcn_combine(const float* __restrict__ HC, const float* __restrict__ acc,
               const float* __restrict__ cnt, const float* __restrict__ bng,
               const float* __restrict__ bnb, const float* __restrict__ bnm,
               const float* __restrict__ bnv, const float* __restrict__ prelu_a,
               const float* __restrict__ xres, float* __restrict__ outf,
               ushort_t* __restrict__ outbf){
  int n = blockIdx.x, f = threadIdx.x;
  float v = HC[(size_t)n * 768 + f];
  #pragma unroll
  for (int r = 0; r < NRELC; ++r){
    float c = cnt[(size_t)r * NN + n];
    v += acc[((size_t)r * NN + n) * 128 + f] / fmaxf(c, 1.0f);
  }
  v = (v - bnm[f]) * rsqrtf(bnv[f] + 1e-5f) * bng[f] + bnb[f];
  float a = prelu_a[0];
  v = (v > 0.0f) ? v : a * v;
  if (xres) v += xres[(size_t)n * 128 + f];
  outf[(size_t)n * 128 + f]  = v;
  outbf[(size_t)n * 128 + f] = f2bf(v);
}

// ------------------------- MLP epilogues -----------------------------------

// relu -> LayerNorm(H) -> *g+b -> bf16 pack (zero-pad to ldout); 1 wave / row
__global__ void __launch_bounds__(128)
k_relu_ln_pack(const float* __restrict__ in, int ldin, int H,
               const float* __restrict__ g, const float* __restrict__ b,
               ushort_t* __restrict__ out, int ldout){
  int row  = blockIdx.x * 4 + (threadIdx.x >> 5);
  int lane = threadIdx.x & 31;
  if (row >= NN) return;
  const float* p = in + (size_t)row * ldin;
  float s = 0.f, s2 = 0.f;
  for (int i = lane; i < H; i += 32){
    float v = fmaxf(p[i], 0.f); s += v; s2 += v * v;
  }
  #pragma unroll
  for (int m = 16; m; m >>= 1){ s += __shfl_xor(s, m, 32); s2 += __shfl_xor(s2, m, 32); }
  float mu = s / H, var = s2 / H - mu * mu, rs = rsqrtf(var + 1e-5f);
  ushort_t* o = out + (size_t)row * ldout;
  for (int i = lane; i < ldout; i += 32){
    float y = 0.f;
    if (i < H){ float v = fmaxf(p[i], 0.f); y = (v - mu) * rs * g[i] + b[i]; }
    o[i] = f2bf(y);
  }
}

// tiny env (64x 8->12->16) and time (6->7->8) MLPs; one block
__global__ void k_env_time(const float* __restrict__ envf, const float* __restrict__ timef, int t,
    const float* ew1, const float* eb1, const float* eg, const float* ebl,
    const float* ew2, const float* eb2,
    const float* tw1, const float* tb1, const float* tg, const float* tbl,
    const float* tw2, const float* tb2,
    float* __restrict__ env_emb, float* __restrict__ time_emb){
  int id = threadIdx.x;
  if (id < NG){
    const float* x = envf + ((size_t)t * NG + id) * 8;
    float h[12]; float s = 0.f, s2 = 0.f;
    for (int j = 0; j < 12; ++j){
      float a = eb1[j];
      for (int k = 0; k < 8; ++k) a += x[k] * ew1[k * 12 + j];
      a = fmaxf(a, 0.f); h[j] = a; s += a; s2 += a * a;
    }
    float mu = s / 12.f, var = s2 / 12.f - mu * mu, rs = rsqrtf(var + 1e-5f);
    for (int j = 0; j < 12; ++j) h[j] = (h[j] - mu) * rs * eg[j] + ebl[j];
    for (int o = 0; o < 16; ++o){
      float a = eb2[o];
      for (int j = 0; j < 12; ++j) a += h[j] * ew2[j * 16 + o];
      env_emb[id * 16 + o] = a;
    }
  } else if (id == NG){
    const float* x = timef + (size_t)t * 6;
    float h[7]; float s = 0.f, s2 = 0.f;
    for (int j = 0; j < 7; ++j){
      float a = tb1[j];
      for (int k = 0; k < 6; ++k) a += x[k] * tw1[k * 7 + j];
      a = fmaxf(a, 0.f); h[j] = a; s += a; s2 += a * a;
    }
    float mu = s / 7.f, var = s2 / 7.f - mu * mu, rs = rsqrtf(var + 1e-5f);
    for (int j = 0; j < 7; ++j) h[j] = (h[j] - mu) * rs * tg[j] + tbl[j];
    for (int o = 0; o < 8; ++o){
      float a = tb2[o];
      for (int j = 0; j < 7; ++j) a += h[j] * tw2[j * 8 + o];
      time_emb[o] = a;
    }
  }
}

// A = bf16[ feats(128) | env_emb[batch](16) | time_emb(8) | pad(8) ]  (NN x 160)
__global__ void k_build_fused(const float* __restrict__ feats, const float* __restrict__ env_emb,
                              const float* __restrict__ time_emb, const int* __restrict__ batch,
                              ushort_t* __restrict__ out){
  int idx = blockIdx.x * 256 + threadIdx.x;
  if (idx >= NN * 160) return;
  int n = idx / 160, c = idx % 160;
  float v = 0.f;
  if (c < 128)       v = feats[(size_t)n * 128 + c];
  else if (c < 144)  v = env_emb[(size_t)batch[n] * 16 + (c - 128)];
  else if (c < 152)  v = time_emb[c - 144];
  out[idx] = f2bf(v);
}

// ------------------------- LSTM + heads ------------------------------------

__global__ void k_lstm_cell(const float* __restrict__ gates, float* __restrict__ h,
                            float* __restrict__ c, ushort_t* __restrict__ hbf){
  int idx = blockIdx.x * 256 + threadIdx.x;
  if (idx >= NN * 128) return;
  int n = idx >> 7, j = idx & 127;
  const float* g = gates + (size_t)n * 512;
  float gi = sigm(g[j]);
  float gf = sigm(g[128 + j]);
  float gg = tanhf(g[256 + j]);
  float go = sigm(g[384 + j]);
  float cn = gf * c[idx] + gi * gg;
  float hn = go * tanhf(cn);
  c[idx] = cn; h[idx] = hn; hbf[idx] = f2bf(hn);
}

// pred[n, t] = relu(H1[n]) . w2 + b2 ; one wave per node
__global__ void __launch_bounds__(128)
k_head_final(const float* __restrict__ H1, const float* __restrict__ w2,
             const float* __restrict__ b2, float* __restrict__ out, int tcol){
  int row  = blockIdx.x * 4 + (threadIdx.x >> 5);
  int lane = threadIdx.x & 31;
  if (row >= NN) return;
  const float* p = H1 + (size_t)row * 64;
  float s = 0.f;
  for (int i = lane; i < 64; i += 32){ float v = fmaxf(p[i], 0.f); s += v * w2[i]; }
  #pragma unroll
  for (int m = 16; m; m >>= 1) s += __shfl_xor(s, m, 32);
  if (lane == 0) out[(size_t)row * TT + tcol] = s + b2[0];
}

// ---------------------------------------------------------------------------
// Host orchestration
// ---------------------------------------------------------------------------
// d_in flat order (setup_inputs insertion order, depth-first through params):
//  0 x_seq  1 env_feats  2 time_feats  3 feat_mean  4 feat_std
//  gen_h0: 5 l0.w_root 6 l0.w_rel 7 l0.bias 8 bn_g 9 bn_b 10 bn_mean 11 bn_var 12 prelu
//          13..20 l1.*   21 fc_w 22 fc_b
//  gen_seq: 23..38 (same layout) 39 fc_w 40 fc_b
//  h0_enc: 41 w1 42 b1 43 ln_g 44 ln_b 45 w2 46 b2
//  env:    47..52   time: 53..58   fusion: 59..64
//  lstm:   65 w_ih 66 w_hh 67 b
//  heads:  68 w1 69 b1 70 w2 71 b2
//  72 edge_index  73 edge_type  74 batch_vec

extern "C" void kernel_launch(void* const* d_in, const int* in_sizes, int n_in,
                              void* d_out, int out_size, void* d_ws, size_t ws_size,
                              hipStream_t stream){
  (void)in_sizes; (void)n_in; (void)out_size; (void)ws_size;
  auto F = [&](int i){ return (const float*)d_in[i]; };
  auto I = [&](int i){ return (const int*)d_in[i]; };
  float* out = (float*)d_out;

  // ---- workspace arena (deterministic layout every call) ----
  char* ws = (char*)d_ws; size_t off = 0;
  auto alloc = [&](size_t bytes)->char*{
    char* p = ws + off; off = (off + bytes + 255) & ~(size_t)255; return p;
  };
  ushort_t* WT_L0[2]; ushort_t* WT_L1[2]; ushort_t* WT_FC[2];
  float* B768_L0[2]; float* B768_L1[2];
  for (int s = 0; s < 2; ++s){
    WT_L0[s] = (ushort_t*)alloc(768 * 32 * 2);
    WT_L1[s] = (ushort_t*)alloc(768 * 128 * 2);
    WT_FC[s] = (ushort_t*)alloc(128 * 128 * 2);
    B768_L0[s] = (float*)alloc(768 * 4);
    B768_L1[s] = (float*)alloc(768 * 4);
  }
  ushort_t* WT_H0W1 = (ushort_t*)alloc(128 * 128 * 2);
  ushort_t* WT_H0W2 = (ushort_t*)alloc(128 * 128 * 2);
  ushort_t* WT_FUS1 = (ushort_t*)alloc(192 * 160 * 2);   // N padded 140 -> 192
  ushort_t* WT_FUS2 = (ushort_t*)alloc(128 * 160 * 2);
  ushort_t* WT_WIH  = (ushort_t*)alloc(512 * 128 * 2);
  ushort_t* WT_WHH  = (ushort_t*)alloc(512 * 128 * 2);
  ushort_t* WT_HEAD = (ushort_t*)alloc((size_t)TT * 64 * 128 * 2);
  float*    B192    = (float*)alloc(192 * 4);

  ushort_t* XBF   = (ushort_t*)alloc((size_t)NN * 32 * 2);
  ushort_t* HBF   = (ushort_t*)alloc((size_t)NN * 128 * 2);
  ushort_t* FBF   = (ushort_t*)alloc((size_t)NN * 160 * 2);
  ushort_t* FBF2  = (ushort_t*)alloc((size_t)NN * 160 * 2);
  ushort_t* FUSBF = (ushort_t*)alloc((size_t)NN * 128 * 2);
  ushort_t* HSBF  = (ushort_t*)alloc((size_t)NN * 128 * 2);
  float* HC    = (float*)alloc((size_t)NN * 768 * 4);   // also hosts LSTM gates (NN*512)
  float* ACC   = (float*)alloc((size_t)NRELC * NN * 128 * 4);
  float* CNT   = (float*)alloc((size_t)NRELC * NN * 4);
  float* HF32A = (float*)alloc((size_t)NN * 128 * 4);
  float* HF32B = (float*)alloc((size_t)NN * 128 * 4);
  float* FEATS = (float*)alloc((size_t)NN * 128 * 4);
  float* G1    = (float*)alloc((size_t)NN * 192 * 4);   // fusion hidden, ld 192
  float* HST   = (float*)alloc((size_t)NN * 128 * 4);
  float* CST   = (float*)alloc((size_t)NN * 128 * 4);
  float* ENVE  = (float*)alloc(NG * 16 * 4);
  float* TIME  = (float*)alloc(8 * 4);
  float* GATES = HC;                                    // lifetime-disjoint alias
  float* FUSED = ACC;                                   // lifetime-disjoint alias
  float* H1    = ACC + (size_t)NN * 128;                // lifetime-disjoint alias

  const int* EI = I(72); const int* ET = I(73); const int* BV = I(74);

  auto gemm = [&](const ushort_t* A, const ushort_t* Bt, const float* bias,
                  float* C, int M, int N, int K, int flags){
    int tiles = (M / 16) * (N / 64);                    // N % 64 == 0 guaranteed
    k_gemm<<<dim3((tiles + 3) / 4), dim3(128), 0, stream>>>(A, Bt, bias, C, M, N, K, flags);
  };
  auto grid = [](int n){ return dim3((n + 255) / 256); };

  // ---- one-time weight packing (bf16, transposed, padded) ----
  for (int s = 0; s < 2; ++s){
    int gb = (s == 0) ? 5 : 23;
    k_pack_rgcnT<<<grid(768 * 32), 256, 0, stream>>>(F(gb + 0), F(gb + 1), WT_L0[s], 32);
    k_pack_rgcnT<<<grid(768 * 128), 256, 0, stream>>>(F(gb + 8), F(gb + 9), WT_L1[s], 128);
    k_pack_wT<<<grid(128 * 128), 256, 0, stream>>>(F(gb + 16), WT_FC[s], 128, 128, 128, 128);
    k_pad_bias<<<grid(768), 256, 0, stream>>>(F(gb + 2), B768_L0[s], 128, 768);
    k_pad_bias<<<grid(768), 256, 0, stream>>>(F(gb + 10), B768_L1[s], 128, 768);
  }
  k_pack_wT<<<grid(128 * 128), 256, 0, stream>>>(F(41), WT_H0W1, 128, 128, 128, 128);
  k_pack_wT<<<grid(128 * 128), 256, 0, stream>>>(F(45), WT_H0W2, 128, 128, 128, 128);
  k_pack_wT<<<grid(192 * 160), 256, 0, stream>>>(F(59), WT_FUS1, 152, 140, 160, 192);
  k_pack_wT<<<grid(128 * 160), 256, 0, stream>>>(F(63), WT_FUS2, 140, 128, 160, 128);
  k_pack_wT<<<grid(512 * 128), 256, 0, stream>>>(F(65), WT_WIH, 128, 512, 128, 512);
  k_pack_wT<<<grid(512 * 128), 256, 0, stream>>>(F(66), WT_WHH, 128, 512, 128, 512);
  k_pack_headT<<<grid(TT * 64 * 128), 256, 0, stream>>>(F(68), WT_HEAD);
  k_pad_bias<<<grid(192), 256, 0, stream>>>(F(60), B192, 140, 192);

  // ---- timestep loop ----
  for (int t = 0; t <= TT; ++t){
    int s  = (t == 0) ? 0 : 1;
    int gb = (t == 0) ? 5 : 23;

    // normalized bf16 node features for this timestep
    k_pack_x<<<grid(NN * 32), 256, 0, stream>>>(
        F(0) + (size_t)t * NN * 32, F(3), F(4), XBF);

    // --- RGCN layer 0 (K=32, N=768 fused root+5 relations) ---
    hipMemsetAsync(ACC, 0, (size_t)NRELC * NN * 128 * 4, stream);
    hipMemsetAsync(CNT, 0, (size_t)NRELC * NN * 4, stream);
    gemm(XBF, WT_L0[s], B768_L0[s], HC, NN, 768, 32, 0);
    k_edge_scatter<<<grid(NE * 32), 256, 0, stream>>>(HC, EI, ET, ACC, CNT);
    k_rgcn_combine<<<NN, 128, 0, stream>>>(HC, ACC, CNT,
        F(gb + 3), F(gb + 4), F(gb + 5), F(gb + 6), F(gb + 7),
        (const float*)nullptr, HF32A, HBF);

    // --- RGCN layer 1 (K=128, residual) ---
    hipMemsetAsync(ACC, 0, (size_t)NRELC * NN * 128 * 4, stream);
    hipMemsetAsync(CNT, 0, (size_t)NRELC * NN * 4, stream);
    gemm(HBF, WT_L1[s], B768_L1[s], HC, NN, 768, 128, 0);
    k_edge_scatter<<<grid(NE * 32), 256, 0, stream>>>(HC, EI, ET, ACC, CNT);
    k_rgcn_combine<<<NN, 128, 0, stream>>>(HC, ACC, CNT,
        F(gb + 11), F(gb + 12), F(gb + 13), F(gb + 14), F(gb + 15),
        HF32A, HF32B, HBF);

    // --- fc: feats = h @ fc_w + fc_b ---
    gemm(HBF, WT_FC[s], F(gb + 17), FEATS, NN, 128, 128, 0);

    if (t == 0){
      // h0 = mlp_enc(feats, h0_enc); c0 = 0
      k_pack_bf<<<grid(NN * 128), 256, 0, stream>>>(FEATS, HBF, NN * 128);
      gemm(HBF, WT_H0W1, F(42), G1, NN, 128, 128, 0);
      k_relu_ln_pack<<<dim3((NN + 3) / 4), 128, 0, stream>>>(G1, 128, 128, F(43), F(44), HBF, 128);
      gemm(HBF, WT_H0W2, F(46), HST, NN, 128, 128, 0);
      k_pack_bf<<<grid(NN * 128), 256, 0, stream>>>(HST, HSBF, NN * 128);
      hipMemsetAsync(CST, 0, (size_t)NN * 128 * 4, stream);
    } else {
      int tc = t - 1;
      // env / time embeddings + fused input
      k_env_time<<<1, 128, 0, stream>>>(F(1), F(2), tc,
          F(47), F(48), F(49), F(50), F(51), F(52),
          F(53), F(54), F(55), F(56), F(57), F(58), ENVE, TIME);
      k_build_fused<<<grid(NN * 160), 256, 0, stream>>>(FEATS, ENVE, TIME, BV, FBF);
      gemm(FBF, WT_FUS1, B192, G1, NN, 192, 160, 0);
      k_relu_ln_pack<<<dim3((NN + 3) / 4), 128, 0, stream>>>(G1, 192, 140, F(61), F(62), FBF2, 160);
      gemm(FBF2, WT_FUS2, F(64), FUSED, NN, 128, 160, 0);
      k_pack_bf<<<grid(NN * 128), 256, 0, stream>>>(FUSED, FUSBF, NN * 128);

      // LSTM step: gates = fused @ w_ih + b + h @ w_hh
      gemm(FUSBF, WT_WIH, F(67), GATES, NN, 512, 128, 0);
      gemm(HSBF,  WT_WHH, nullptr, GATES, NN, 512, 128, 1);
      k_lstm_cell<<<grid(NN * 128), 256, 0, stream>>>(GATES, HST, CST, HSBF);

      // head for this timestep
      gemm(HSBF, WT_HEAD + (size_t)tc * 64 * 128, F(69) + (size_t)tc * 64,
           H1, NN, 64, 128, 0);
      k_head_final<<<dim3((NN + 3) / 4), 128, 0, stream>>>(
          H1, F(70) + (size_t)tc * 64, F(71) + tc, out, tc);
    }
  }
}
